// SpikeFP32MultiHeadAttention_43860206027262
// MI455X (gfx1250) — matrix operand order repaired and checked by reading, not compiled, not run on hardware
//
#include <hip/hip_runtime.h>
#include <hip/hip_bf16.h>

typedef _Float16 v16h __attribute__((ext_vector_type(16)));
typedef _Float16 h8   __attribute__((ext_vector_type(8)));
typedef float    v8f  __attribute__((ext_vector_type(8)));
typedef float    f4   __attribute__((ext_vector_type(4)));

static constexpr int Bsz = 4, Sq = 2048, Em = 1024, Hh = 16, Dh = 64;

__device__ __forceinline__ v8f wmma_f16(v16h a, v16h b, v8f c) {
    return __builtin_amdgcn_wmma_f32_16x16x32_f16(false, a, false, b, (short)0, c, false, false);
}

__device__ __forceinline__ h8 cvt_h8(f4 a, f4 b) {
    h8 r;
#pragma unroll
    for (int i = 0; i < 4; ++i) { r[i] = (_Float16)a[i]; r[4 + i] = (_Float16)b[i]; }
    return r;
}

__device__ __forceinline__ v16h make_v16(h8 lo, h8 hi) {
    v16h r;
#pragma unroll
    for (int i = 0; i < 8; ++i) { r[i] = lo[i]; r[8 + i] = hi[i]; }
    return r;
}

// CDNA5 async DMA: copy 16 bytes/lane from global to LDS (ASYNCcnt-tracked).
__device__ __forceinline__ void async_copy16(unsigned lds_off, const void* gptr) {
    asm volatile("global_load_async_to_lds_b128 %0, %1, off"
                 :
                 : "v"(lds_off), "v"((unsigned long long)(uintptr_t)gptr)
                 : "memory");
}
// Wait until this wave's ASYNCcnt <= N (async loads complete in order, so
// waiting to <= batch_size guarantees the previous tile's batch has landed).
template <int N>
__device__ __forceinline__ void wait_async() {
    asm volatile("s_wait_asynccnt %0" ::"i"(N) : "memory");
}

// A-fragment (16x32 f16, M x K), f16 source: lane holds row (caller selects),
// two 8-half chunks at k = c0.. and k = c0+16.. with c0 = 8*(lane/16).
__device__ __forceinline__ v16h frag_a_h(const _Float16* rowk, int lane) {
    int c0 = ((lane >> 4) << 3);
    h8 lo = *(const h8*)(rowk + c0);
    h8 hi = *(const h8*)(rowk + c0 + 16);
    return make_v16(lo, hi);
}

// A-fragment from f32 source (converted on load).
__device__ __forceinline__ v16h frag_a_f32(const float* rowk, int lane) {
    int c0 = ((lane >> 4) << 3);
    const f4* p = (const f4*)(rowk + c0);
    const f4* q = (const f4*)(rowk + c0 + 16);
    f4 x0 = p[0], x1 = p[1], x2 = q[0], x3 = q[1];
    return make_v16(cvt_h8(x0, x1), cvt_h8(x2, x3));
}

// B-fragment (32x16 f16, K x N), f16 source: lane holds column (caller selects),
// two 8-half chunks at k = c0.. and k = c0+16.. with c0 = 8*(lane/16).
__device__ __forceinline__ v16h frag_b_h(const _Float16* colk, int lane) {
    int off = ((lane >> 4) << 3);
    const h8* p = (const h8*)(colk + off);
    return make_v16(p[0], p[2]);
}

// B-fragment from f32 source (converted on load).
__device__ __forceinline__ v16h frag_b_f32(const float* colk, int lane) {
    int off = ((lane >> 4) << 3);
    const f4* p = (const f4*)(colk + off);
    f4 x0 = p[0], x1 = p[1], x2 = p[4], x3 = p[5];
    return make_v16(cvt_h8(x0, x1), cvt_h8(x2, x3));
}

// ---------------------------------------------------------------------------
// NT GEMM: C[M,N] = A[M,K] * Bw[N,K]^T   (einsum 'mk,nk->mn')
// block = 128 threads (4 waves, 2x2), block tile 64x64, K step 32.
// Double-buffered LDS staging via GLOBAL_LOAD_ASYNC_TO_LDS_B128: tile t+1's
// DMA overlaps tile t's WMMAs; s_wait_asynccnt <= BATCH fences only the
// previous batch. f32->f16 conversion folded into fragment loads from LDS.
// ---------------------------------------------------------------------------
template <typename TA, typename TC>
__global__ __launch_bounds__(128) void gemm_nt_wmma(const TA* __restrict__ A,
                                                    const float* __restrict__ Bw,
                                                    TC* __restrict__ C,
                                                    int M, int N, int K) {
    // Per buffer: A tile f32 64x32 pitch 36 (9216B) or f16 64x32 pitch 48 (6144B)
    __shared__ __align__(16) char smemA[2][64 * 36 * 4];
    __shared__ __align__(16) char smemB[2][64 * 36 * 4];

    const int lane = threadIdx.x & 31;
    const int wave = threadIdx.x >> 5;
    const int wm = wave >> 1, wn = wave & 1;
    const int m0 = blockIdx.y * 64, n0 = blockIdx.x * 64;
    const int srow = threadIdx.x >> 1, shalf = threadIdx.x & 1;

    // Per-wave async batch: B = 4 x b128; A = 4 (f32) or 2 (f16) x b128.
    constexpr int BATCH = (sizeof(TA) == 4) ? 8 : 6;

    auto stage = [&](int kk, int bufi) {
        const float* gb = Bw + (size_t)(n0 + srow) * K + kk + shalf * 16;
        unsigned lb = (unsigned)(uintptr_t)((float*)smemB[bufi] + srow * 36 + shalf * 16);
#pragma unroll
        for (int q = 0; q < 4; ++q) async_copy16(lb + q * 16, gb + q * 4);
        if constexpr (sizeof(TA) == 4) {
            const TA* ga = A + (size_t)(m0 + srow) * K + kk + shalf * 16;
            unsigned la = (unsigned)(uintptr_t)((float*)smemA[bufi] + srow * 36 + shalf * 16);
#pragma unroll
            for (int q = 0; q < 4; ++q) async_copy16(la + q * 16, ga + q * 4);
        } else {
            const TA* ga = A + (size_t)(m0 + srow) * K + kk + shalf * 16;
            unsigned la = (unsigned)(uintptr_t)((_Float16*)smemA[bufi] + srow * 48 + shalf * 16);
#pragma unroll
            for (int q = 0; q < 2; ++q) async_copy16(la + q * 16, ga + q * 8);
        }
    };

    v8f c[2][2] = {};
    const int T = K / 32;

    stage(0, 0);  // prologue: tile 0 -> buffer 0

    for (int t = 0; t < T; ++t) {
        if (t + 1 < T) {
            stage((t + 1) * 32, (t + 1) & 1);  // overlap next tile's DMA
            wait_async<BATCH>();               // oldest batch (tile t) landed
        } else {
            wait_async<0>();
        }
        __syncthreads();  // publish tile t across waves

        const int bufc = t & 1;
        const float* AsF = (const float*)smemA[bufc];
        const _Float16* AsH = (const _Float16*)smemA[bufc];
        const float* BsF = (const float*)smemB[bufc];

        v16h af[2], bf[2];
#pragma unroll
        for (int i = 0; i < 2; ++i) {
            int row = wm * 32 + i * 16 + (lane & 15);
            if constexpr (sizeof(TA) == 4)
                af[i] = frag_a_f32(AsF + row * 36, lane);
            else
                af[i] = frag_a_h(AsH + row * 48, lane);
        }
#pragma unroll
        for (int j = 0; j < 2; ++j)
            bf[j] = frag_b_f32(BsF + (wn * 32 + j * 16 + (lane & 15)) * 36, lane);
#pragma unroll
        for (int i = 0; i < 2; ++i)
#pragma unroll
            for (int j = 0; j < 2; ++j)
                c[i][j] = wmma_f16(af[i], bf[j], c[i][j]);

        __syncthreads();  // reads done; buffer may be re-staged next iteration
    }

#pragma unroll
    for (int i = 0; i < 2; ++i)
#pragma unroll
        for (int j = 0; j < 2; ++j)
#pragma unroll
            for (int r = 0; r < 8; ++r) {
                int row = m0 + wm * 32 + i * 16 + r + ((lane >> 4) << 3);
                int col = n0 + wn * 32 + j * 16 + (lane & 15);
                C[(size_t)row * N + col] = (TC)c[i][j][r];
            }
}

// ---------------------------------------------------------------------------
// RoPE + permute: Qp/Kp/Vp are [B,S,E] f16 -> Qh,Kh [B,H,S,D] f16 (roped),
// Vt [B,H,D,S] f16 (transposed so PV WMMA B-fragments are contiguous).
// One thread per (b,s,h,i) with i in [0,32) handling the (i, i+32) pair.
// ---------------------------------------------------------------------------
__global__ __launch_bounds__(256) void rope_permute_kernel(
    const _Float16* __restrict__ Qp, const _Float16* __restrict__ Kp,
    const _Float16* __restrict__ Vp, const long long* __restrict__ pos,
    _Float16* __restrict__ Qh, _Float16* __restrict__ Kh, _Float16* __restrict__ Vt) {
    int idx = blockIdx.x * blockDim.x + threadIdx.x;
    int i = idx & 31; int r = idx >> 5;
    int h = r & (Hh - 1); r >>= 4;
    int s = r & (Sq - 1); int b = r >> 11;

    const float ln_base = 9.210340371976184f;  // ln(10000)
    float freq = __expf(-((float)(2 * i) * (1.0f / (float)Dh)) * ln_base);
    float ang = (float)pos[s] * freq;
    float sn, cs;
    __sincosf(ang, &sn, &cs);

    size_t src = ((size_t)b * Sq + s) * Em + h * Dh + i;
    int bh = b * Hh + h;
    size_t dst = ((size_t)bh * Sq + s) * Dh + i;

    float q1 = (float)Qp[src], q2 = (float)Qp[src + 32];
    Qh[dst]      = (_Float16)(q1 * cs - q2 * sn);
    Qh[dst + 32] = (_Float16)(q2 * cs + q1 * sn);

    float k1 = (float)Kp[src], k2 = (float)Kp[src + 32];
    Kh[dst]      = (_Float16)(k1 * cs - k2 * sn);
    Kh[dst + 32] = (_Float16)(k2 * cs + k1 * sn);

    Vt[((size_t)bh * Dh + i) * Sq + s]      = Vp[src];
    Vt[((size_t)bh * Dh + i + 32) * Sq + s] = Vp[src + 32];
}

// ---------------------------------------------------------------------------
// Causal flash attention. Grid: (S/128, B*H). 8 waves/block; each wave owns a
// 16-row Q tile and streams KV in 32-column tiles with online softmax.
// ---------------------------------------------------------------------------
__global__ __launch_bounds__(256) void flash_attn_kernel(
    const _Float16* __restrict__ Qh, const _Float16* __restrict__ Kh,
    const _Float16* __restrict__ Vt, _Float16* __restrict__ O) {
    __shared__ _Float16 Ps[8][16 * 48];
    const int lane = threadIdx.x & 31;
    const int w = threadIdx.x >> 5;
    const int q0 = blockIdx.x * 128 + w * 16;
    const int bh = blockIdx.y;
    const int b = bh >> 4, h = bh & 15;
    const float scale = 0.125f;  // 1/sqrt(64)

    const _Float16* qbase = Qh + (size_t)bh * Sq * Dh;
    const _Float16* kbase = Kh + (size_t)bh * Sq * Dh;
    const _Float16* vbase = Vt + (size_t)bh * Dh * Sq;

    v16h qf[2];
#pragma unroll
    for (int ks = 0; ks < 2; ++ks)
        qf[ks] = frag_a_h(qbase + (size_t)(q0 + (lane & 15)) * Dh + ks * 32, lane);

    float mi[8], li[8];
    v8f o[4] = {};
#pragma unroll
    for (int r = 0; r < 8; ++r) { mi[r] = -1e30f; li[r] = 0.0f; }

    const int qlast = q0 + 15;
    const int rowq = q0 + ((lane >> 4) << 3);

    for (int kv0 = 0; kv0 <= qlast; kv0 += 32) {
        // Prefetch next KV tile into the cache hierarchy (global_prefetch_b8).
        if (kv0 + 32 <= qlast) {
            __builtin_prefetch(kbase + (size_t)(kv0 + 32 + (lane & 15)) * Dh, 0, 1);
            __builtin_prefetch(vbase + (size_t)(lane & 15) * Sq + kv0 + 32, 0, 1);
        }

        v8f s0 = {}, s1 = {};
#pragma unroll
        for (int ks = 0; ks < 2; ++ks) {
            v16h kf0 = frag_b_h(kbase + (size_t)(kv0 + (lane & 15)) * Dh + ks * 32, lane);
            v16h kf1 = frag_b_h(kbase + (size_t)(kv0 + 16 + (lane & 15)) * Dh + ks * 32, lane);
            s0 = wmma_f16(qf[ks], kf0, s0);
            s1 = wmma_f16(qf[ks], kf1, s1);
        }

        const int nc0 = kv0 + (lane & 15), nc1 = nc0 + 16;
        float al[8], p0[8], p1[8];
#pragma unroll
        for (int r = 0; r < 8; ++r) {
            float a = s0[r] * scale;
            float c = s1[r] * scale;
            if (nc0 > rowq + r) a = -1e30f;
            if (nc1 > rowq + r) c = -1e30f;
            float t = fmaxf(a, c);
            t = fmaxf(t, __shfl_xor(t, 1, 32));
            t = fmaxf(t, __shfl_xor(t, 2, 32));
            t = fmaxf(t, __shfl_xor(t, 4, 32));
            t = fmaxf(t, __shfl_xor(t, 8, 32));
            float mnew = fmaxf(mi[r], t);
            al[r] = __expf(mi[r] - mnew);
            p0[r] = __expf(a - mnew);
            p1[r] = __expf(c - mnew);
            float rs = p0[r] + p1[r];
            rs += __shfl_xor(rs, 1, 32);
            rs += __shfl_xor(rs, 2, 32);
            rs += __shfl_xor(rs, 4, 32);
            rs += __shfl_xor(rs, 8, 32);
            li[r] = li[r] * al[r] + rs;
            mi[r] = mnew;
        }
#pragma unroll
        for (int j = 0; j < 4; ++j)
#pragma unroll
            for (int r = 0; r < 8; ++r) o[j][r] *= al[r];

        // C-layout -> A-layout transpose of P through per-wave LDS tile.
        _Float16* pb = Ps[w];
#pragma unroll
        for (int r = 0; r < 8; ++r) {
            int row = r + ((lane >> 4) << 3);
            pb[row * 48 + (lane & 15)]      = (_Float16)p0[r];
            pb[row * 48 + 16 + (lane & 15)] = (_Float16)p1[r];
        }
        v16h pf = frag_a_h(pb + (lane & 15) * 48, lane);  // same-wave DS: in-order
#pragma unroll
        for (int j = 0; j < 4; ++j) {
            v16h vf = frag_b_h(vbase + (size_t)(j * 16 + (lane & 15)) * Sq + kv0, lane);
            o[j] = wmma_f16(pf, vf, o[j]);
        }
    }

    const size_t obase = ((size_t)b * Sq) * Em + (size_t)h * Dh;
#pragma unroll
    for (int r = 0; r < 8; ++r) {
        float inv = 1.0f / li[r];
        int row = q0 + r + ((lane >> 4) << 3);
#pragma unroll
        for (int j = 0; j < 4; ++j)
            O[obase + (size_t)row * Em + j * 16 + (lane & 15)] = (_Float16)(o[j][r] * inv);
    }
}

// ---------------------------------------------------------------------------
extern "C" void kernel_launch(void* const* d_in, const int* in_sizes, int n_in,
                              void* d_out, int out_size, void* d_ws, size_t ws_size,
                              hipStream_t stream) {
    const float* query = (const float*)d_in[0];
    const float* key   = (const float*)d_in[1];
    const float* value = (const float*)d_in[2];
    const float* Wq    = (const float*)d_in[3];
    const float* Wk    = (const float*)d_in[4];
    const float* Wv    = (const float*)d_in[5];
    const float* Wo    = (const float*)d_in[6];
    const long long* positions = (const long long*)d_in[7];
    float* out = (float*)d_out;

    const int M = Bsz * Sq;  // 8192
    const size_t SZ16 = (size_t)M * Em * sizeof(_Float16);  // 16 MB
    char* ws = (char*)d_ws;
    _Float16* Qp = (_Float16*)(ws + 0 * SZ16);
    _Float16* Kp = (_Float16*)(ws + 1 * SZ16);
    _Float16* Vp = (_Float16*)(ws + 2 * SZ16);
    _Float16* Qh = (_Float16*)(ws + 3 * SZ16);
    _Float16* Kh = (_Float16*)(ws + 4 * SZ16);
    _Float16* Vt = (_Float16*)(ws + 5 * SZ16);
    _Float16* Ao = Qp;  // Qp is dead after rope_permute; reuse for attention out

    dim3 gg(Em / 64, M / 64);  // (16, 128)
    gemm_nt_wmma<float, _Float16><<<gg, 128, 0, stream>>>(query, Wq, Qp, M, Em, Em);
    gemm_nt_wmma<float, _Float16><<<gg, 128, 0, stream>>>(key,   Wk, Kp, M, Em, Em);
    gemm_nt_wmma<float, _Float16><<<gg, 128, 0, stream>>>(value, Wv, Vp, M, Em, Em);

    int rope_threads = Bsz * Sq * Hh * 32;
    rope_permute_kernel<<<rope_threads / 256, 256, 0, stream>>>(Qp, Kp, Vp, positions,
                                                                Qh, Kh, Vt);

    flash_attn_kernel<<<dim3(Sq / 128, Bsz * Hh), 256, 0, stream>>>(Qh, Kh, Vt, Ao);

    gemm_nt_wmma<_Float16, float><<<gg, 128, 0, stream>>>(Ao, Wo, out, M, Em, Em);
}
